// GraphEmbedding_1099511628325
// MI455X (gfx1250) — matrix operand order repaired and checked by reading, not compiled, not run on hardware
//
#include <hip/hip_runtime.h>
#include <hip/hip_bf16.h>
#include <math.h>

// ---------------------------------------------------------------------------
// Problem constants (match reference)
// ---------------------------------------------------------------------------
#define NL 12000
#define ND 6000
#define FDIM 256
#define EL 384000
#define ED 192000
#define ATT_NORM 0.0625f   /* 1/sqrt(256) */

typedef __attribute__((ext_vector_type(16))) _Float16 v16h;
typedef __attribute__((ext_vector_type(8)))  float    v8f;

// ---------------------------------------------------------------------------
// Small utility kernels
// ---------------------------------------------------------------------------
__global__ void k_fill_f32(float* p, int n, float v) {
  int i = blockIdx.x * blockDim.x + threadIdx.x;
  if (i < n) p[i] = v;
}
__global__ void k_fill_u32(unsigned* p, int n, unsigned v) {
  int i = blockIdx.x * blockDim.x + threadIdx.x;
  if (i < n) p[i] = v;
}

__device__ __forceinline__ float waveSum(float v) {
  #pragma unroll
  for (int o = 16; o; o >>= 1) v += __shfl_down(v, o, 32);
  return v;
}

// ordered-uint encoding so atomicMax(unsigned) == float max
__device__ __forceinline__ unsigned f2o(float f) {
  unsigned u = __float_as_uint(f);
  return (u & 0x80000000u) ? ~u : (u | 0x80000000u);
}
__device__ __forceinline__ float o2f(unsigned u) {
  return (u & 0x80000000u) ? __uint_as_float(u & 0x7fffffffu) : __uint_as_float(~u);
}
#define ORDERED_NEG_INF 0x007FFFFFu  /* f2o(-inf) */

// ---------------------------------------------------------------------------
// Block-cooperative WMMA GEMM: C[M,N] = A[M,K] * B + bias.
//   BT=false: B is [K,N] row-major.   BT=true: B is [N,K] row-major (A*B^T).
// Block = 256 threads (8 waves) computing a 64x64 output tile.
// Per K-step (BK=32): stage A panel (64x32) and B panel (32x64) in LDS as f16,
// each wave computes a 16x32 strip = 2 x v_wmma_f32_16x16x32_f16, reusing its
// A fragment across both. Out-of-range blocks use clamped loads (block-uniform
// branch -> EXEC stays all-1 around WMMA) and per-element store guards.
// K must be a multiple of 32 (K=256 everywhere here).
// ---------------------------------------------------------------------------
#define BM 64
#define BN 64
#define BK 32

template <bool BT>
__global__ __launch_bounds__(256)
void k_gemm_wmma(const float* __restrict__ A, const float* __restrict__ B,
                 float* __restrict__ C, const float* __restrict__ bias,
                 int M, int N, int K) {
  __shared__ _Float16 Ah[BM][BK];   // [row][k]
  __shared__ _Float16 Bh[BN][BK];   // [col][k]  (k contiguous for both NN/NT)

  const int t    = threadIdx.x;
  const int lane = t & 31;
  const int wave = t >> 5;

  const int blocksN = (N + BN - 1) / BN;
  const int bm = blockIdx.x / blocksN;
  const int bn = blockIdx.x - bm * blocksN;
  const int rowBase = bm * BM;
  const int colBase = bn * BN;
  const bool fullM = (rowBase + BM) <= M;
  const bool fullN = (colBase + BN) <= N;

  // wave -> (row tile 0..3, col-tile-pair 0..1)
  const int rw = wave >> 1;
  const int cw = wave & 1;
  const int r  = lane & 15;
  const int hi = lane >> 4;
  const int akoff = hi * 16;   // A frag: K 0..15 (lo) vs 16..31 (hi)
  const int bkoff = hi * 16;   // B frag: K 0..15 (lo) vs 16..31 (hi)

  v8f acc0 = {}, acc1 = {};

  // cooperative-fill indices
  const int aRow  = t >> 2;            // 0..63
  const int aKseg = (t & 3) * 8;       // 0,8,16,24
  const int gARow = fullM ? (rowBase + aRow) : min(rowBase + aRow, M - 1);

  for (int kb = 0; kb < K; kb += BK) {
    // ---- fill A panel: each thread converts 8 consecutive K floats ----
    {
      const float* ap = A + (size_t)gARow * K + kb + aKseg;
      #pragma unroll
      for (int j = 0; j < 8; ++j) Ah[aRow][aKseg + j] = (_Float16)ap[j];
    }
    // ---- fill B panel into Bh[col][k] ----
    if (BT) {
      const int c    = t >> 2;         // 0..63
      const int kseg = (t & 3) * 8;
      const int gc   = fullN ? (colBase + c) : min(colBase + c, N - 1);
      const float* bp = B + (size_t)gc * K + kb + kseg;
      #pragma unroll
      for (int j = 0; j < 8; ++j) Bh[c][kseg + j] = (_Float16)bp[j];
    } else {
      const int k    = t >> 3;         // 0..31
      const int cseg = (t & 7) * 8;    // 0..56
      if (fullN) {
        const float* bp = B + (size_t)(kb + k) * N + colBase + cseg;
        #pragma unroll
        for (int j = 0; j < 8; ++j) Bh[cseg + j][k] = (_Float16)bp[j];
      } else {
        #pragma unroll
        for (int j = 0; j < 8; ++j) {
          int gc = min(colBase + cseg + j, N - 1);
          Bh[cseg + j][k] = (_Float16)B[(size_t)(kb + k) * N + gc];
        }
      }
    }
    __syncthreads();

    // ---- fragments from LDS (contiguous k order: element j -> k = 16*hi + j, same for A and B) ----
    v16h a;
    {
      const _Float16* ap0 = &Ah[rw * 16 + r][akoff];
      const _Float16* ap1 = ap0 + 8;
      #pragma unroll
      for (int j = 0; j < 8; ++j) { a[j] = ap0[j]; a[j + 8] = ap1[j]; }
    }
    v16h b0, b1;
    {
      const _Float16* bp0 = &Bh[cw * 32 + r][bkoff];
      const _Float16* bp1 = &Bh[cw * 32 + 16 + r][bkoff];
      #pragma unroll
      for (int j = 0; j < 16; ++j) { b0[j] = bp0[j]; b1[j] = bp1[j]; }
    }
    acc0 = __builtin_amdgcn_wmma_f32_16x16x32_f16(
        false, a, false, b0, (short)0, acc0, false, false);
    acc1 = __builtin_amdgcn_wmma_f32_16x16x32_f16(
        false, a, false, b1, (short)0, acc1, false, false);
    __syncthreads();
  }

  // ---- epilogue: C/D layout VGPR v -> row v (+8 for hi half), col = lane&15
  const int col0 = colBase + cw * 32 + r;
  const int col1 = col0 + 16;
  const float bv0 = (bias && col0 < N) ? bias[col0] : 0.0f;
  const float bv1 = (bias && col1 < N) ? bias[col1] : 0.0f;
  #pragma unroll
  for (int v = 0; v < 8; ++v) {
    const int row = rowBase + rw * 16 + v + hi * 8;
    if (row < M) {
      if (col0 < N) C[(size_t)row * N + col0] = acc0[v] + bv0;
      if (col1 < N) C[(size_t)row * N + col1] = acc1[v] + bv1;
    }
  }
}

// ---------------------------------------------------------------------------
// GCN kernels
// ---------------------------------------------------------------------------
__global__ void k_deg_scatter(int E, const int* __restrict__ dst,
                              const float* __restrict__ w, float* __restrict__ deg) {
  int e = blockIdx.x * blockDim.x + threadIdx.x;
  if (e < E) atomicAdd(&deg[dst[e]], w[e]);
}

__global__ void k_rsqrt(int n, const float* __restrict__ deg, float* __restrict__ dinv) {
  int i = blockIdx.x * blockDim.x + threadIdx.x;
  if (i < n) dinv[i] = rsqrtf(fmaxf(deg[i], 1e-12f));
}

// out = dinv[i]^2 * h   (self-loop contribution, weight 1)
__global__ void k_gcn_self(int n, const float* __restrict__ dinv,
                           const float* __restrict__ h, float* __restrict__ out) {
  int idx = blockIdx.x * blockDim.x + threadIdx.x;
  if (idx < n * FDIM) {
    int node = idx >> 8;
    float dv = dinv[node];
    out[idx] = dv * dv * h[idx];
  }
}

// one wave per edge: out[d] += dinv[s]*w*dinv[d] * h[s]
__global__ __launch_bounds__(256)
void k_gcn_edge(int E, const int* __restrict__ src, const int* __restrict__ dst,
                const float* __restrict__ w, const float* __restrict__ dinv,
                const float* __restrict__ h, float* __restrict__ out) {
  int wave = blockIdx.x * (blockDim.x >> 5) + (threadIdx.x >> 5);
  if (wave >= E) return;
  int lane = threadIdx.x & 31;
  int s = src[wave], d = dst[wave];
  const float* hs = h + (size_t)s * FDIM;
  float* od = out + (size_t)d * FDIM;
  __builtin_prefetch(hs + lane * 8, 0, 0);   // global_prefetch_b8 on gathered row
  float norm = dinv[s] * w[wave] * dinv[d];
  #pragma unroll
  for (int j = lane; j < FDIM; j += 32) atomicAdd(&od[j], norm * hs[j]);
}

__global__ void k_bias_relu(int n, const float* __restrict__ in,
                            const float* __restrict__ b, float* __restrict__ out) {
  int idx = blockIdx.x * blockDim.x + threadIdx.x;
  if (idx < n * FDIM) out[idx] = fmaxf(in[idx] + b[idx & (FDIM - 1)], 0.0f);
}

// ---------------------------------------------------------------------------
// GAT kernels
// ---------------------------------------------------------------------------
// per node: as = h . att_src, ad = h . att_dst  (one wave per node)
__global__ __launch_bounds__(256)
void k_node_dots(int n, const float* __restrict__ h,
                 const float* __restrict__ asrc, const float* __restrict__ adst,
                 float* __restrict__ as_, float* __restrict__ ad_) {
  int wave = blockIdx.x * (blockDim.x >> 5) + (threadIdx.x >> 5);
  if (wave >= n) return;
  int lane = threadIdx.x & 31;
  const float* row = h + (size_t)wave * FDIM;
  float s0 = 0.f, s1 = 0.f;
  #pragma unroll
  for (int j = lane; j < FDIM; j += 32) {
    float v = row[j];
    s0 += v * asrc[j];
    s1 += v * adst[j];
  }
  s0 = waveSum(s0);
  s1 = waveSum(s1);
  if (lane == 0) { as_[wave] = s0; ad_[wave] = s1; }
}

// single-block 256-element dot product -> out[0]
__global__ void k_dot256(const float* __restrict__ a, const float* __restrict__ b,
                         float* __restrict__ out) {
  __shared__ float sm[8];
  int t = threadIdx.x;
  float v = a[t] * b[t];
  v = waveSum(v);
  if ((t & 31) == 0) sm[t >> 5] = v;
  __syncthreads();
  if (t == 0) {
    float s = 0.f;
    #pragma unroll
    for (int i = 0; i < 8; ++i) s += sm[i];
    out[0] = s;
  }
}

// grid sum of w -> atomicAdd into out[0] (out must be pre-zeroed)
__global__ void k_sum(int n, const float* __restrict__ w, float* __restrict__ out) {
  int i = blockIdx.x * blockDim.x + threadIdx.x;
  float v = (i < n) ? w[i] : 0.f;
  v = waveSum(v);
  if ((threadIdx.x & 31) == 0) atomicAdd(out, v);
}

__device__ __forceinline__ float leaky02(float a) { return a > 0.f ? a : 0.2f * a; }

__global__ void k_gat_edge_max(int E, const int* __restrict__ src, const int* __restrict__ dst,
                               const float* __restrict__ w, const float* __restrict__ as_,
                               const float* __restrict__ ad_, const float* __restrict__ cE,
                               unsigned* __restrict__ amaxu) {
  int e = blockIdx.x * blockDim.x + threadIdx.x;
  if (e >= E) return;
  float a = leaky02(as_[src[e]] + ad_[dst[e]] + w[e] * cE[0]);
  atomicMax(&amaxu[dst[e]], f2o(a));
}

__global__ void k_gat_node_max(int n, const float* __restrict__ as_, const float* __restrict__ ad_,
                               const float* __restrict__ wsum, const float* __restrict__ cE,
                               float invE, unsigned* __restrict__ amaxu) {
  int i = blockIdx.x * blockDim.x + threadIdx.x;
  if (i >= n) return;
  float a = leaky02(as_[i] + ad_[i] + (wsum[0] * invE) * cE[0]);
  atomicMax(&amaxu[i], f2o(a));
}

__global__ void k_gat_edge_denom(int E, const int* __restrict__ src, const int* __restrict__ dst,
                                 const float* __restrict__ w, const float* __restrict__ as_,
                                 const float* __restrict__ ad_, const float* __restrict__ cE,
                                 const unsigned* __restrict__ amaxu, float* __restrict__ denom) {
  int e = blockIdx.x * blockDim.x + threadIdx.x;
  if (e >= E) return;
  int d = dst[e];
  float a = leaky02(as_[src[e]] + ad_[d] + w[e] * cE[0]);
  atomicAdd(&denom[d], __expf(a - o2f(amaxu[d])));
}

__global__ void k_gat_node_denom(int n, const float* __restrict__ as_, const float* __restrict__ ad_,
                                 const float* __restrict__ wsum, const float* __restrict__ cE,
                                 float invE, const unsigned* __restrict__ amaxu,
                                 float* __restrict__ denom) {
  int i = blockIdx.x * blockDim.x + threadIdx.x;
  if (i >= n) return;
  float a = leaky02(as_[i] + ad_[i] + (wsum[0] * invE) * cE[0]);
  atomicAdd(&denom[i], __expf(a - o2f(amaxu[i])));
}

// out = alpha_self * h   (self-loop contribution of GAT aggregation)
__global__ void k_gat_self_out(int n, const float* __restrict__ as_, const float* __restrict__ ad_,
                               const float* __restrict__ wsum, const float* __restrict__ cE,
                               float invE, const unsigned* __restrict__ amaxu,
                               const float* __restrict__ denom, const float* __restrict__ h,
                               float* __restrict__ out) {
  int idx = blockIdx.x * blockDim.x + threadIdx.x;
  if (idx >= n * FDIM) return;
  int i = idx >> 8;
  float a  = leaky02(as_[i] + ad_[i] + (wsum[0] * invE) * cE[0]);
  float al = __expf(a - o2f(amaxu[i])) / (denom[i] + 1e-16f);
  out[idx] = al * h[idx];
}

// one wave per edge: out[d] += alpha_e * h[s]
__global__ __launch_bounds__(256)
void k_gat_edge_out(int E, const int* __restrict__ src, const int* __restrict__ dst,
                    const float* __restrict__ w, const float* __restrict__ as_,
                    const float* __restrict__ ad_, const float* __restrict__ cE,
                    const unsigned* __restrict__ amaxu, const float* __restrict__ denom,
                    const float* __restrict__ h, float* __restrict__ out) {
  int wave = blockIdx.x * (blockDim.x >> 5) + (threadIdx.x >> 5);
  if (wave >= E) return;
  int lane = threadIdx.x & 31;
  int s = src[wave], d = dst[wave];
  const float* hs = h + (size_t)s * FDIM;
  float* od = out + (size_t)d * FDIM;
  __builtin_prefetch(hs + lane * 8, 0, 0);
  float a  = leaky02(as_[s] + ad_[d] + w[wave] * cE[0]);
  float al = __expf(a - o2f(amaxu[d])) / (denom[d] + 1e-16f);
  #pragma unroll
  for (int j = lane; j < FDIM; j += 32) atomicAdd(&od[j], al * hs[j]);
}

// ---------------------------------------------------------------------------
// Layer attention: per-node 2x2 attention + Conv1d(2->1) combine.
// One wave per node.
// ---------------------------------------------------------------------------
__global__ __launch_bounds__(256)
void k_layeratt(int n, const float* __restrict__ Q1, const float* __restrict__ Q2,
                const float* __restrict__ K1, const float* __restrict__ K2,
                const float* __restrict__ V1, const float* __restrict__ V2,
                const float* __restrict__ cw, const float* __restrict__ cb,
                float* __restrict__ out) {
  int wave = blockIdx.x * (blockDim.x >> 5) + (threadIdx.x >> 5);
  if (wave >= n) return;
  int lane = threadIdx.x & 31;
  size_t base = (size_t)wave * FDIM;

  float s00 = 0.f, s01 = 0.f, s10 = 0.f, s11 = 0.f;
  #pragma unroll
  for (int j = lane; j < FDIM; j += 32) {
    float q1 = Q1[base + j], q2 = Q2[base + j];
    float k1 = K1[base + j], k2 = K2[base + j];
    s00 += q1 * k1; s01 += q1 * k2;
    s10 += q2 * k1; s11 += q2 * k2;
  }
  s00 = __shfl(waveSum(s00), 0, 32) * ATT_NORM;
  s01 = __shfl(waveSum(s01), 0, 32) * ATT_NORM;
  s10 = __shfl(waveSum(s10), 0, 32) * ATT_NORM;
  s11 = __shfl(waveSum(s11), 0, 32) * ATT_NORM;

  // softmax over l (axis=1) for each m (column-wise)
  float m0 = fmaxf(s00, s10);
  float e00 = __expf(s00 - m0), e10 = __expf(s10 - m0);
  float inv0 = 1.0f / (e00 + e10);
  float al00 = e00 * inv0, al10 = e10 * inv0;

  float m1 = fmaxf(s01, s11);
  float e01 = __expf(s01 - m1), e11 = __expf(s11 - m1);
  float inv1 = 1.0f / (e01 + e11);
  float al01 = e01 * inv1, al11 = e11 * inv1;

  float cw0 = cw[0], cw1 = cw[1], cbv = cb[0];
  float c1 = cw0 * al00 + cw1 * al10;   // weight on V1 (m=0)
  float c2 = cw0 * al01 + cw1 * al11;   // weight on V2 (m=1)

  #pragma unroll
  for (int j = lane; j < FDIM; j += 32)
    out[base + j] = c1 * V1[base + j] + c2 * V2[base + j] + cbv;
}

// ---------------------------------------------------------------------------
// Host-side orchestration
// ---------------------------------------------------------------------------
static inline int ceil_div(int a, int b) { return (a + b - 1) / b; }

static void launch_gemm_nn(const float* A, const float* B, float* C, const float* bias,
                           int M, int N, int K, hipStream_t s) {
  int blocks = ceil_div(M, BM) * ceil_div(N, BN);
  k_gemm_wmma<false><<<blocks, 256, 0, s>>>(A, B, C, bias, M, N, K);
}
static void launch_gemm_nt(const float* A, const float* B, float* C, const float* bias,
                           int M, int N, int K, hipStream_t s) {
  int blocks = ceil_div(M, BM) * ceil_div(N, BN);
  k_gemm_wmma<true><<<blocks, 256, 0, s>>>(A, B, C, bias, M, N, K);
}

struct Scratch {
  float *H, *F1, *FA, *F2, *X1, *X2, *X3, *X4;
  float *deg, *dinv, *as_, *ad_, *denom;
  unsigned *amaxu;
  float *cE, *wsum;
};

static void run_branch(const float* x, const int* edges_flat, const float* w, int E, int n,
                       const float* const* P /*18 branch params*/, float* fea_out,
                       const Scratch& S, hipStream_t s) {
  const int* src = edges_flat;
  const int* dst = edges_flat + E;
  const float invE = 1.0f / (float)E;

  const float* W1 = P[0];  const float* b1 = P[1];
  const float* Wg = P[2];  const float* att_src = P[3]; const float* att_dst = P[4];
  const float* lin_e = P[5]; const float* att_e = P[6]; const float* bg = P[7];
  const float* W2 = P[8];  const float* b2 = P[9];
  const float* Wq = P[10]; const float* bq = P[11];
  const float* Wk = P[12]; const float* bk = P[13];
  const float* Wv = P[14]; const float* bv = P[15];
  const float* cw = P[16]; const float* cb = P[17];

  const int nf = n * FDIM;
  const int gN  = ceil_div(n, 256);
  const int gNF = ceil_div(nf, 256);
  const int gE  = ceil_div(E, 256);
  const int gEw = ceil_div(E, 8);    // wave-per-edge
  const int gNw = ceil_div(n, 8);    // wave-per-node

  // ---- GCN normalization (shared by gcn1 and gcn2; ww identical) ----
  k_fill_f32<<<gN, 256, 0, s>>>(S.deg, n, 1.0f);                  // self-loop weight 1
  k_deg_scatter<<<gE, 256, 0, s>>>(E, dst, w, S.deg);
  k_rsqrt<<<gN, 256, 0, s>>>(n, S.deg, S.dinv);

  // ---- GCN1: f1 = relu(norm-scatter(x @ W1) + b1) ----
  launch_gemm_nn(x, W1, S.H, nullptr, n, FDIM, FDIM, s);
  k_gcn_self<<<gNF, 256, 0, s>>>(n, S.dinv, S.H, S.F1);
  k_gcn_edge<<<gEw, 256, 0, s>>>(E, src, dst, w, S.dinv, S.H, S.F1);
  k_bias_relu<<<gNF, 256, 0, s>>>(n, S.F1, b1, S.F1);

  // ---- GAT: fa = relu(gat(f1) + bg) ----
  launch_gemm_nn(S.F1, Wg, S.H, nullptr, n, FDIM, FDIM, s);
  k_node_dots<<<gNw, 256, 0, s>>>(n, S.H, att_src, att_dst, S.as_, S.ad_);
  k_dot256<<<1, 256, 0, s>>>(lin_e, att_e, S.cE);
  k_fill_f32<<<1, 32, 0, s>>>(S.wsum, 1, 0.0f);
  k_sum<<<gE, 256, 0, s>>>(E, w, S.wsum);
  k_fill_u32<<<gN, 256, 0, s>>>(S.amaxu, n, ORDERED_NEG_INF);
  k_gat_edge_max<<<gE, 256, 0, s>>>(E, src, dst, w, S.as_, S.ad_, S.cE, S.amaxu);
  k_gat_node_max<<<gN, 256, 0, s>>>(n, S.as_, S.ad_, S.wsum, S.cE, invE, S.amaxu);
  k_fill_f32<<<gN, 256, 0, s>>>(S.denom, n, 0.0f);
  k_gat_edge_denom<<<gE, 256, 0, s>>>(E, src, dst, w, S.as_, S.ad_, S.cE, S.amaxu, S.denom);
  k_gat_node_denom<<<gN, 256, 0, s>>>(n, S.as_, S.ad_, S.wsum, S.cE, invE, S.amaxu, S.denom);
  k_gat_self_out<<<gNF, 256, 0, s>>>(n, S.as_, S.ad_, S.wsum, S.cE, invE, S.amaxu, S.denom, S.H, S.FA);
  k_gat_edge_out<<<gEw, 256, 0, s>>>(E, src, dst, w, S.as_, S.ad_, S.cE, S.amaxu, S.denom, S.H, S.FA);
  k_bias_relu<<<gNF, 256, 0, s>>>(n, S.FA, bg, S.FA);

  // ---- GCN2: f2 = relu(norm-scatter(fa @ W2) + b2) ----
  launch_gemm_nn(S.FA, W2, S.H, nullptr, n, FDIM, FDIM, s);
  k_gcn_self<<<gNF, 256, 0, s>>>(n, S.dinv, S.H, S.F2);
  k_gcn_edge<<<gEw, 256, 0, s>>>(E, src, dst, w, S.dinv, S.H, S.F2);
  k_bias_relu<<<gNF, 256, 0, s>>>(n, S.F2, b2, S.F2);

  // ---- Layer attention over [f1, f2] ----
  launch_gemm_nn(S.F1, Wq, S.H,  bq, n, FDIM, FDIM, s);  // Q1 (reuse H)
  launch_gemm_nn(S.F2, Wq, S.FA, bq, n, FDIM, FDIM, s);  // Q2 (reuse FA)
  launch_gemm_nn(S.F1, Wk, S.X1, bk, n, FDIM, FDIM, s);  // K1
  launch_gemm_nn(S.F2, Wk, S.X2, bk, n, FDIM, FDIM, s);  // K2
  launch_gemm_nn(S.F1, Wv, S.X3, bv, n, FDIM, FDIM, s);  // V1
  launch_gemm_nn(S.F2, Wv, S.X4, bv, n, FDIM, FDIM, s);  // V2
  k_layeratt<<<gNw, 256, 0, s>>>(n, S.H, S.FA, S.X1, S.X2, S.X3, S.X4, cw, cb, fea_out);
}

extern "C" void kernel_launch(void* const* d_in, const int* in_sizes, int n_in,
                              void* d_out, int out_size, void* d_ws, size_t ws_size,
                              hipStream_t stream) {
  (void)in_sizes; (void)n_in; (void)out_size; (void)ws_size;

  // Inputs in setup_inputs() dict (insertion) order; params flattened recursively.
  const float* x_lnc  = (const float*)d_in[0];
  const float* x_dis  = (const float*)d_in[1];
  const int*   ll_e   = (const int*)d_in[2];
  const float* ll_w   = (const float*)d_in[3];
  const int*   dd_e   = (const int*)d_in[4];
  const float* dd_w   = (const float*)d_in[5];
  const float* Plnc[18];
  const float* Pdis[18];
  for (int i = 0; i < 18; ++i) {
    Plnc[i] = (const float*)d_in[6 + i];
    Pdis[i] = (const float*)d_in[24 + i];
  }

  // Output layout: [NL*ND scores][NL*F lnc_fea][ND*F dis_fea]
  float* out_scores = (float*)d_out;
  float* lnc_fea = out_scores + (size_t)NL * ND;
  float* dis_fea = lnc_fea + (size_t)NL * FDIM;

  // Workspace layout (floats). 8 node buffers sized for NL + per-node scalars.
  const size_t NODE = (size_t)NL * FDIM;
  float* W = (float*)d_ws;
  Scratch S;
  S.H  = W + 0 * NODE;  S.F1 = W + 1 * NODE;  S.FA = W + 2 * NODE;  S.F2 = W + 3 * NODE;
  S.X1 = W + 4 * NODE;  S.X2 = W + 5 * NODE;  S.X3 = W + 6 * NODE;  S.X4 = W + 7 * NODE;
  float* scal = W + 8 * NODE;
  S.deg   = scal;
  S.dinv  = scal + 1 * NL;
  S.as_   = scal + 2 * NL;
  S.ad_   = scal + 3 * NL;
  S.denom = scal + 4 * NL;
  S.amaxu = (unsigned*)(scal + 5 * NL);
  S.cE    = scal + 6 * NL;
  S.wsum  = S.cE + 1;

  // Two branches sequentially (scratch reused), then final score GEMM (NT).
  run_branch(x_lnc, ll_e, ll_w, EL, NL, Plnc, lnc_fea, S, stream);
  run_branch(x_dis, dd_e, dd_w, ED, ND, Pdis, dis_fea, S, stream);
  launch_gemm_nt(lnc_fea, dis_fea, out_scores, nullptr, NL, ND, FDIM, stream);
}
